// Network_52304111730743
// MI455X (gfx1250) — matrix operand order repaired and checked by reading, not compiled, not run on hardware
//
#include <hip/hip_runtime.h>
#include <math.h>

typedef _Float16 h16;
typedef __attribute__((ext_vector_type(16))) _Float16 v16h;
typedef __attribute__((ext_vector_type(8)))  _Float16 v8h;
typedef __attribute__((ext_vector_type(8)))  float    v8f;

#define BB   16
#define TT   512
#define FF   17
#define KK   20
#define CP   32            // channel dim padded to 32 (zero-filled 20..31)
#define TTILE 32
#define NTIL  16           // 512/32
#define RI   (TTILE+6)     // 38 input rows (halo 3)
#define R0   (TTILE+4)     // 36 act0 rows (halo 2)
#define R1   (TTILE+2)     // 34 act1 rows (halo 1)
#define FP   (FF+2)        // 19, f padded by 1 each side
#define NPOS1 (R1*FF)      // 578
#define MT1   37
#define NPOS2 (TTILE*FF)   // 544
#define MT2   34
#define FEAT  (FF*KK)      // 340
#define FRAME 30
#define NFR   (TT-FRAME+1) // 483
#define EPSF  1e-12f

// LDS carve sizes (halves / floats)
#define BFRAG_H (2*9*2*32*16)          // 18432: [layer][tap][ntile][lane][16]
#define ACT0_H  (R0*FP*CP)             // 21888
#define ACT1_H  (R1*FP*CP)             // 20672
#define INP_F   (RI*FP)                // 722
#define WB_F    264                    // w0[180], b0[20], b1[32], b2[32]
#define TOWER_LDS ((BFRAG_H+ACT0_H+ACT1_H)*2 + (INP_F+WB_F)*4)
#define CORR_LDS  ((FRAME*FEAT*2 + 8)*4)

union AV { v16h v; v8h h[2]; };

__global__ __launch_bounds__(256) void tower_kernel(
    const float* __restrict__ s, const float* __restrict__ x,
    const float* __restrict__ w0, const float* __restrict__ b0,
    const float* __restrict__ w1, const float* __restrict__ b1,
    const float* __restrict__ w2, const float* __restrict__ b2,
    float* __restrict__ Sout, float* __restrict__ Xout)
{
    extern __shared__ char smemraw[];
    h16*   Bfrag = (h16*)smemraw;                // [2][9][2][32][16]
    h16*   act0  = Bfrag + BFRAG_H;              // [R0][FP][CP]
    h16*   act1  = act0 + ACT0_H;                // [R1][FP][CP]
    float* inp   = (float*)(act1 + ACT1_H);      // [RI][FP]
    float* wb    = inp + INP_F;                  // w0 | b0 | b1pad | b2pad

    const int tid = threadIdx.x;
    const int bx  = blockIdx.x;
    const int tileIdx = bx & 15;
    const int which   = (bx >> 4) & 1;
    const int b       = bx >> 5;
    const int t0      = tileIdx * TTILE;
    const float* in  = which ? x : s;
    float*       out = which ? Xout : Sout;

    // ---- zero LDS activations (incl. channel pad) + input pad ----
    {
        unsigned int* za = (unsigned int*)act0;
        for (int i = tid; i < (ACT0_H + ACT1_H) / 2; i += 256) za[i] = 0u;
        for (int i = tid; i < INP_F; i += 256) inp[i] = 0.f;
    }
    // ---- build per-tap B fragments in exact WMMA lane layout ----
    // index = (((l*9+tap)*2+nt)*32 + lane)*16 + e ; B layout: K = 8g + e (e<8), 16 + 8g + (e-8) (e>=8)
    for (int i = tid; i < BFRAG_H; i += 256) {
        int e = i & 15, lane = (i >> 4) & 31, nt = (i >> 9) & 1;
        int lt = i >> 10;                 // l*9 + tap
        int tap = lt % 9, l = lt / 9;
        int g = lane >> 4;
        int n = nt * 16 + (lane & 15);
        int k = (e < 8) ? (g * 8 + e) : (16 + g * 8 + (e - 8));
        float v = 0.f;
        if (k < 20 && n < 20) v = (l ? w2 : w1)[(tap * 20 + k) * 20 + n];
        Bfrag[i] = (h16)v;
    }
    // ---- conv0 weights + biases ----
    for (int i = tid; i < WB_F; i += 256) {
        float v;
        if (i < 180)      v = w0[i];
        else if (i < 200) v = b0[i - 180];
        else if (i < 232) { int n = i - 200; v = (n < 20) ? b1[n] : 0.f; }
        else              { int n = i - 232; v = (n < 20) ? b2[n] : 0.f; }
        wb[i] = v;
    }
    __syncthreads();

    // ---- load padded input tile ----
    for (int i = tid; i < RI * FF; i += 256) {
        int r = i / FF, f = i % FF;
        int tg = t0 - 3 + r;
        if (tg >= 0 && tg < TT)
            inp[r * FP + f + 1] = in[((size_t)b * TT + tg) * FF + f];
    }
    __syncthreads();

    // ---- layer0 (1 -> 20), plain VALU, K_red = 9 ----
    for (int i = tid; i < R0 * FF * KK; i += 256) {
        int k = i % 20, q = i / 20;
        int fo = q % FF, ra = q / FF;
        int tg = t0 - 2 + ra;
        if (tg < 0 || tg >= TT) continue;       // stays zero
        float acc = wb[180 + k];
        #pragma unroll
        for (int dt = 0; dt < 3; ++dt)
            #pragma unroll
            for (int df = 0; df < 3; ++df)
                acc += inp[(ra + dt) * FP + fo + df] * wb[(dt * 3 + df) * 20 + k];
        act0[(ra * FP + fo + 1) * CP + k] = (h16)(acc > 0.f ? acc : 0.f);
    }
    __syncthreads();

    const int lane = tid & 31;
    const int wv   = tid >> 5;
    const int m_a  = lane & 15;
    const int g    = lane >> 4;
    const int n_d  = lane & 15;
    const int gofs = g * 8;              // half offset of this lane-group's K run

    // ---- layer1 (20 -> 20): 9 tap-GEMMs, WMMA 16x16x32 f16: act0 -> act1 ----
    for (int mt = wv; mt < MT1; mt += 8) {
        int p = mt * 16 + m_a; if (p >= NPOS1) p = 0;
        int trow = p / FF, fo = p % FF;
        int baseA = (trow * FP + fo) * CP + gofs;
        float bias0 = wb[200 + n_d];
        float bias1 = wb[200 + 16 + n_d];
        v8f acc0, acc1;
        #pragma unroll
        for (int r = 0; r < 8; ++r) { acc0[r] = bias0; acc1[r] = bias1; }
        #pragma unroll
        for (int dt = 0; dt < 3; ++dt) {
            #pragma unroll
            for (int df = 0; df < 3; ++df) {
                const int tap = dt * 3 + df;
                const int tofs = (dt * FP + df) * CP;
                AV a;
                a.h[0] = *(const v8h*)&act0[baseA + tofs];        // K = 8g+0..7
                a.h[1] = *(const v8h*)&act0[baseA + tofs + 16];   // K = 16+8g+0..7
                const v16h bf0 = *(const v16h*)&Bfrag[((0 * 9 + tap) * 2 + 0) * 512 + lane * 16];
                const v16h bf1 = *(const v16h*)&Bfrag[((0 * 9 + tap) * 2 + 1) * 512 + lane * 16];
                acc0 = __builtin_amdgcn_wmma_f32_16x16x32_f16(false, a.v, false, bf0, (short)0, acc0, false, false);
                acc1 = __builtin_amdgcn_wmma_f32_16x16x32_f16(false, a.v, false, bf1, (short)0, acc1, false, false);
            }
        }
        #pragma unroll
        for (int r = 0; r < 8; ++r) {
            int prow = mt * 16 + r + 8 * g;
            if (prow < NPOS1) {
                int tr = prow / FF, frp = prow % FF;
                float v0 = acc0[r]; v0 = v0 > 0.f ? v0 : 0.f;
                act1[(tr * FP + frp + 1) * CP + n_d] = (h16)v0;
                if (n_d + 16 < 20) {
                    float v1 = acc1[r]; v1 = v1 > 0.f ? v1 : 0.f;
                    act1[(tr * FP + frp + 1) * CP + 16 + n_d] = (h16)v1;
                }
            }
        }
    }
    __syncthreads();

    // ---- layer2 (20 -> 20): act1 -> global (B,T,340) f32 ----
    for (int mt = wv; mt < MT2; mt += 8) {
        int p = mt * 16 + m_a; if (p >= NPOS2) p = 0;
        int trow = p / FF, fo = p % FF;
        int baseA = (trow * FP + fo) * CP + gofs;
        float bias0 = wb[232 + n_d];
        float bias1 = wb[232 + 16 + n_d];
        v8f acc0, acc1;
        #pragma unroll
        for (int r = 0; r < 8; ++r) { acc0[r] = bias0; acc1[r] = bias1; }
        #pragma unroll
        for (int dt = 0; dt < 3; ++dt) {
            #pragma unroll
            for (int df = 0; df < 3; ++df) {
                const int tap = dt * 3 + df;
                const int tofs = (dt * FP + df) * CP;
                AV a;
                a.h[0] = *(const v8h*)&act1[baseA + tofs];
                a.h[1] = *(const v8h*)&act1[baseA + tofs + 16];
                const v16h bf0 = *(const v16h*)&Bfrag[((1 * 9 + tap) * 2 + 0) * 512 + lane * 16];
                const v16h bf1 = *(const v16h*)&Bfrag[((1 * 9 + tap) * 2 + 1) * 512 + lane * 16];
                acc0 = __builtin_amdgcn_wmma_f32_16x16x32_f16(false, a.v, false, bf0, (short)0, acc0, false, false);
                acc1 = __builtin_amdgcn_wmma_f32_16x16x32_f16(false, a.v, false, bf1, (short)0, acc1, false, false);
            }
        }
        #pragma unroll
        for (int r = 0; r < 8; ++r) {
            int prow = mt * 16 + r + 8 * g;
            if (prow < NPOS2) {
                int tr = prow / FF, frp = prow % FF;
                int tg = t0 + tr;
                size_t rowb = ((size_t)b * TT + tg) * FEAT + frp * 20;
                float v0 = acc0[r]; v0 = v0 > 0.f ? v0 : 0.f;
                out[rowb + n_d] = v0;
                if (n_d + 16 < 20) {
                    float v1 = acc1[r]; v1 = v1 > 0.f ? v1 : 0.f;
                    out[rowb + 16 + n_d] = v1;
                }
            }
        }
    }
}

// One block per (batch, frame): both 30x340 windows in LDS, double normalization, dot.
__global__ __launch_bounds__(256) void corr_kernel(
    const float* __restrict__ S, const float* __restrict__ X,
    float* __restrict__ corr)
{
    extern __shared__ char smemraw[];
    float* fS  = (float*)smemraw;          // 30*340
    float* fX  = fS + FRAME * FEAT;        // 30*340
    float* red = fX + FRAME * FEAT;        // 8

    const int bx  = blockIdx.x;
    const int fr  = bx % NFR;
    const int b   = bx / NFR;
    const int tid = threadIdx.x;
    const int lane = tid & 31, wv = tid >> 5;

    for (int i = tid; i < FRAME * FEAT; i += 256) {
        int r = i / FEAT, j = i % FEAT;
        size_t gidx = ((size_t)b * TT + fr + r) * FEAT + j;
        fS[i] = S[gidx];
        fX[i] = X[gidx];
    }
    __syncthreads();

    float* arrs[2] = { fS, fX };

    // stage 1: per-feature over 30 rows
    #pragma unroll
    for (int a = 0; a < 2; ++a) {
        float* fA = arrs[a];
        for (int j = tid; j < FEAT; j += 256) {
            float s1 = 0.f;
            for (int r = 0; r < FRAME; ++r) s1 += fA[r * FEAT + j];
            float mu = s1 * (1.f / FRAME);
            float ss = 0.f;
            for (int r = 0; r < FRAME; ++r) { float y = fA[r * FEAT + j] - mu; ss += y * y; }
            float inv = 1.f / (sqrtf(ss) + EPSF);
            for (int r = 0; r < FRAME; ++r) fA[r * FEAT + j] = (fA[r * FEAT + j] - mu) * inv;
        }
    }
    __syncthreads();

    // stage 2: per-row over 340 features, one wave per row
    #pragma unroll
    for (int a = 0; a < 2; ++a) {
        float* fA = arrs[a];
        for (int r = wv; r < FRAME; r += 8) {
            float s1 = 0.f;
            for (int j = lane; j < FEAT; j += 32) s1 += fA[r * FEAT + j];
            for (int o = 16; o > 0; o >>= 1) s1 += __shfl_down(s1, o, 32);
            float mu = __shfl(s1, 0, 32) * (1.f / FEAT);
            float ss = 0.f;
            for (int j = lane; j < FEAT; j += 32) { float z = fA[r * FEAT + j] - mu; ss += z * z; }
            for (int o = 16; o > 0; o >>= 1) ss += __shfl_down(ss, o, 32);
            float inv = 1.f / (sqrtf(__shfl(ss, 0, 32)) + EPSF);
            for (int j = lane; j < FEAT; j += 32) fA[r * FEAT + j] = (fA[r * FEAT + j] - mu) * inv;
        }
    }
    __syncthreads();

    float d = 0.f;
    for (int i = tid; i < FRAME * FEAT; i += 256) d += fS[i] * fX[i];
    for (int o = 16; o > 0; o >>= 1) d += __shfl_down(d, o, 32);
    if (lane == 0) red[wv] = d;
    __syncthreads();
    if (tid == 0) {
        float t = 0.f;
        for (int i = 0; i < 8; ++i) t += red[i];
        corr[b * NFR + fr] = t * (1.f / FRAME);
    }
}

__global__ __launch_bounds__(256) void mean_kernel(
    const float* __restrict__ corr, float* __restrict__ out)
{
    __shared__ float red[8];
    const int b = blockIdx.x, tid = threadIdx.x;
    float s = 0.f;
    for (int i = tid; i < NFR; i += 256) s += corr[b * NFR + i];
    for (int o = 16; o > 0; o >>= 1) s += __shfl_down(s, o, 32);
    if ((tid & 31) == 0) red[tid >> 5] = s;
    __syncthreads();
    if (tid == 0) {
        float t = 0.f;
        for (int i = 0; i < 8; ++i) t += red[i];
        out[b] = t / (float)NFR;
    }
}

extern "C" void kernel_launch(void* const* d_in, const int* in_sizes, int n_in,
                              void* d_out, int out_size, void* d_ws, size_t ws_size,
                              hipStream_t stream) {
    const float* s  = (const float*)d_in[0];
    const float* x  = (const float*)d_in[1];
    const float* w0 = (const float*)d_in[2];
    const float* b0 = (const float*)d_in[3];
    const float* w1 = (const float*)d_in[4];
    const float* b1 = (const float*)d_in[5];
    const float* w2 = (const float*)d_in[6];
    const float* b2 = (const float*)d_in[7];

    float* Sbuf = (float*)d_ws;                        // 16*512*340
    float* Xbuf = Sbuf + (size_t)BB * TT * FEAT;       // 16*512*340
    float* cbuf = Xbuf + (size_t)BB * TT * FEAT;       // 16*483

    tower_kernel<<<BB * 2 * NTIL, 256, TOWER_LDS, stream>>>(
        s, x, w0, b0, w1, b1, w2, b2, Sbuf, Xbuf);
    corr_kernel<<<BB * NFR, 256, CORR_LDS, stream>>>(Sbuf, Xbuf, cbuf);
    mean_kernel<<<BB, 256, 0, stream>>>(cbuf, (float*)d_out);
}
